// HCoN_75299366633945
// MI455X (gfx1250) — matrix operand order repaired and checked by reading, not compiled, not run on hardware
//
#include <hip/hip_runtime.h>
#include <hip/hip_bf16.h>

// ---------------------------------------------------------------------------
// HCoN forward for MI455X (gfx1250): bf16 WMMA GEMMs with fp32 accumulate.
// Streaming-bound on the fp32 adjacency matrices (~1.15 GB); fp32->bf16
// conversion happens in registers (v_perm_b32) while staging into
// double-buffered LDS, so no extra HBM traffic and one barrier per K-tile.
// All matrix math goes through v_wmma_f32_16x16x32_bf16.
// ---------------------------------------------------------------------------

typedef __attribute__((ext_vector_type(16))) __bf16 v16bf;
typedef __attribute__((ext_vector_type(8)))  float  v8f;

#define TILE_M   128
#define TILE_N   64
#define TILE_K   64
#define NTHREADS 256
#define LDA_P    34   // LDS row stride in dwords (32 bf16-pairs + 2 pad, even)
#define LDB_P    34
#define A_BUF    (TILE_M * LDA_P)
#define B_BUF    (TILE_N * LDB_P)

__device__ __forceinline__ unsigned short f2bf(float f) {
  return (unsigned short)((__float_as_uint(f) + 0x8000u) >> 16);
}
// two fp32 -> packed bf16 pair: bias both, merge high halves with one v_perm_b32
__device__ __forceinline__ unsigned int pack2(float lo, float hi) {
  unsigned int ul = __float_as_uint(lo) + 0x8000u;
  unsigned int uh = __float_as_uint(hi) + 0x8000u;
  return __builtin_amdgcn_perm(uh, ul, 0x07060302u);
}

// Stage one 128x64 A tile and 64x64 B tile (fp32 -> bf16) into LDS buffers.
__device__ __forceinline__ void stage_tile(
    const float* __restrict__ A, int lda,
    const float* __restrict__ B, int ldb, int transB,
    int row0, int col0, int k0,
    unsigned int* bufA, unsigned int* bufB, int tid)
{
  // A: 2048 float4 quads, b128 loads, b64 LDS stores.
#pragma unroll
  for (int i = 0; i < 8; ++i) {
    int idx = tid + i * NTHREADS;       // 0..2047
    int m   = idx >> 4;
    int k4  = idx & 15;
    const float4 f = *(const float4*)(A + (size_t)(row0 + m) * lda + (size_t)(k0 + (k4 << 2)));
    uint2 p;
    p.x = pack2(f.x, f.y);
    p.y = pack2(f.z, f.w);
    *(uint2*)&bufA[m * LDA_P + (k4 << 1)] = p;
  }
  unsigned short* bufBh = (unsigned short*)bufB;
  if (!transB) {
    // B: 1024 float4 quads along n, transposed scatter into [n][k].
#pragma unroll
    for (int i = 0; i < 4; ++i) {
      int idx = tid + i * NTHREADS;     // 0..1023
      int n4  = idx & 15;
      int k   = idx >> 4;
      const float4 f = *(const float4*)(B + (size_t)(k0 + k) * ldb + (size_t)(col0 + (n4 << 2)));
      unsigned short* p = bufBh + (n4 << 2) * (2 * LDB_P) + k;
      p[0 * 2 * LDB_P] = f2bf(f.x);
      p[1 * 2 * LDB_P] = f2bf(f.y);
      p[2 * 2 * LDB_P] = f2bf(f.z);
      p[3 * 2 * LDB_P] = f2bf(f.w);
    }
  } else {
    // B is y2^T: scalar gather (only used for the K=64 h_hat GEMM).
#pragma unroll
    for (int i = 0; i < 16; ++i) {
      int idx = tid + i * NTHREADS;     // 0..4095
      int n   = idx & 63;
      int k   = idx >> 6;
      bufBh[n * (2 * LDB_P) + k] = f2bf(B[(size_t)(col0 + n) * ldb + (size_t)(k0 + k)]);
    }
  }
}

// acc += A[row0:+128, :K] * B[:K, col0:+64], double-buffered K loop.
__device__ __forceinline__ void gemm_phase(
    const float* __restrict__ A, int lda,
    const float* __restrict__ B, int ldb, int K, int transB,
    int row0, int col0, unsigned int* ldsA, unsigned int* ldsB,
    v8f (&acc)[2][2])
{
  const int tid  = threadIdx.x;
  const int lane = tid & 31;
  const int wv   = tid >> 5;
  const int wm   = wv & 3;    // wave row block (x32)
  const int wn   = wv >> 2;   // wave col block (x32)
  const int lm   = lane & 15;
  const int lh   = lane >> 4;

  const int nt = K / TILE_K;
  __syncthreads();                       // prior readers of buf0 are done
  stage_tile(A, lda, B, ldb, transB, row0, col0, 0, ldsA, ldsB, tid);

  for (int it = 0; it < nt; ++it) {
    __syncthreads();                     // stores of buf[it&1] visible
    const int cur = it & 1;
    if (it + 1 < nt)                     // prefetch next tile into other buffer
      stage_tile(A, lda, B, ldb, transB, row0, col0, (it + 1) * TILE_K,
                 ldsA + ((it + 1) & 1) * A_BUF, ldsB + ((it + 1) & 1) * B_BUF, tid);

    unsigned int* bufA = ldsA + cur * A_BUF;
    unsigned int* bufB = ldsB + cur * B_BUF;
#pragma unroll
    for (int s = 0; s < 2; ++s) {        // two K=32 sub-steps
      union { unsigned int u[8]; v16bf v; } af[2], bf[2];
#pragma unroll
      for (int t = 0; t < 2; ++t) {
        const unsigned int* ar = &bufA[(wm * 32 + t * 16 + lm) * LDA_P + s * 16];
        const unsigned int* br = &bufB[(wn * 32 + t * 16 + lm) * LDB_P + s * 16];
#pragma unroll
        for (int v = 0; v < 8; ++v) {
          // A 16x32 bf16 layout: kpair = (v/4)*8 + (lane/16)*4 + (v%4)
          af[t].u[v] = ar[((v >> 2) << 3) + (lh << 2) + (v & 3)];
          // B 32x16 bf16 layout: kpair = (v/4)*8 + (lane/16)*4 + (v%4)
          bf[t].u[v] = br[((v >> 2) << 3) + (lh << 2) + (v & 3)];
        }
      }
#pragma unroll
      for (int tm = 0; tm < 2; ++tm)
#pragma unroll
        for (int tn = 0; tn < 2; ++tn)
          acc[tm][tn] = __builtin_amdgcn_wmma_f32_16x16x32_bf16(
              false, af[tm].v, false, bf[tn].v,
              (short)0, acc[tm][tn], false, false);
    }
  }
}

// C = epi( s1*(A1@B1 + bias1) + s2*(A2@B2 + bias2) )
// epi: 0=none, 1=leaky-relu(0.2), 2=sigmoid.  scale_ptr!=null -> s1=*p, s2=1-*p.
__global__ void __launch_bounds__(NTHREADS)
hcon_dual_gemm(const float* __restrict__ A1, int lda1,
               const float* __restrict__ B1, int ldb1, int K1,
               const float* __restrict__ A2, int lda2,
               const float* __restrict__ B2, int ldb2, int K2,
               const float* __restrict__ bias1, const float* __restrict__ bias2,
               const float* __restrict__ scale_ptr, float c1, float c2,
               float* __restrict__ C, int ldc, int transB, int epi)
{
  __shared__ unsigned int ldsA[2 * A_BUF];
  __shared__ unsigned int ldsB[2 * B_BUF];

  const int row0 = blockIdx.y * TILE_M;
  const int col0 = blockIdx.x * TILE_N;

  v8f acc1[2][2] = {};
  v8f acc2[2][2] = {};

  gemm_phase(A1, lda1, B1, ldb1, K1, transB, row0, col0, ldsA, ldsB, acc1);
  if (K2 > 0)
    gemm_phase(A2, lda2, B2, ldb2, K2, transB, row0, col0, ldsA, ldsB, acc2);

  float s1 = c1, s2 = c2;
  if (scale_ptr) { float a = *scale_ptr; s1 = a; s2 = 1.0f - a; }

  const int tid  = threadIdx.x;
  const int lane = tid & 31;
  const int wv   = tid >> 5;
  const int wm   = wv & 3;
  const int wn   = wv >> 2;
  const int lm   = lane & 15;
  const int lh   = lane >> 4;

#pragma unroll
  for (int tn = 0; tn < 2; ++tn) {
    int col = col0 + wn * 32 + tn * 16 + lm;
    float b = 0.0f;
    if (bias1) b += s1 * bias1[col];
    if (bias2) b += s2 * bias2[col];
#pragma unroll
    for (int tm = 0; tm < 2; ++tm) {
      int rbase = row0 + wm * 32 + tm * 16 + lh * 8;   // C/D layout: VGPR r -> M=r (+8 hi half)
#pragma unroll
      for (int r = 0; r < 8; ++r) {
        float v = s1 * acc1[tm][tn][r] + s2 * acc2[tm][tn][r] + b;
        if (epi == 1)      v = (v >= 0.0f) ? v : 0.2f * v;
        else if (epi == 2) v = 1.0f / (1.0f + __expf(-v));
        C[(size_t)(rbase + r) * ldc + col] = v;
      }
    }
  }
}

// Row-wise log_softmax for [rows x 64]: one wave per row, 2 elements per lane.
__global__ void __launch_bounds__(256)
hcon_log_softmax64(const float* __restrict__ X, float* __restrict__ out)
{
  int row  = blockIdx.x * 8 + (threadIdx.x >> 5);
  int lane = threadIdx.x & 31;
  const float* x = X + (size_t)row * 64;
  float v0 = x[lane], v1 = x[lane + 32];
  float m = fmaxf(v0, v1);
#pragma unroll
  for (int off = 16; off > 0; off >>= 1) m = fmaxf(m, __shfl_xor(m, off, 32));
  float s = __expf(v0 - m) + __expf(v1 - m);
#pragma unroll
  for (int off = 16; off > 0; off >>= 1) s += __shfl_xor(s, off, 32);
  float ls = __logf(s) + m;
  out[(size_t)row * 64 + lane]      = v0 - ls;
  out[(size_t)row * 64 + lane + 32] = v1 - ls;
}

extern "C" void kernel_launch(void* const* d_in, const int* in_sizes, int n_in,
                              void* d_out, int out_size, void* d_ws, size_t ws_size,
                              hipStream_t stream) {
  (void)in_sizes; (void)n_in; (void)out_size; (void)ws_size;
  const float* hx1   = (const float*)d_in[0];
  const float* hx2   = (const float*)d_in[1];
  const float* x0    = (const float*)d_in[2];
  const float* hy1   = (const float*)d_in[3];
  const float* hy2   = (const float*)d_in[4];
  const float* y0    = (const float*)d_in[5];
  const float* alpha = (const float*)d_in[6];
  const float* beta  = (const float*)d_in[7];
  const float* Wx1a  = (const float*)d_in[8];
  const float* bx1a  = (const float*)d_in[9];
  const float* Wx1b  = (const float*)d_in[10];
  const float* bx1b  = (const float*)d_in[11];
  const float* Wx2a  = (const float*)d_in[12];
  const float* bx2a  = (const float*)d_in[13];
  const float* Wx2b  = (const float*)d_in[14];
  const float* bx2b  = (const float*)d_in[15];
  const float* Wy1a  = (const float*)d_in[16];
  const float* by1a  = (const float*)d_in[17];
  const float* Wy1b  = (const float*)d_in[18];
  const float* by1b  = (const float*)d_in[19];
  const float* Wy2a  = (const float*)d_in[20];
  const float* by2a  = (const float*)d_in[21];
  const float* Wy2b  = (const float*)d_in[22];
  const float* by2b  = (const float*)d_in[23];

  const int Nx = 8192, Ny = 4096, Fx = 512, Fy = 256, D1 = 256, C = 64;

  float* ws = (float*)d_ws;
  size_t off = 0;
  float* Px1a = ws + off; off += (size_t)Nx * D1;
  float* Px1b = ws + off; off += (size_t)Ny * D1;
  float* Py1a = ws + off; off += (size_t)Ny * D1;
  float* Py1b = ws + off; off += (size_t)Nx * D1;
  float* x1b  = ws + off; off += (size_t)Nx * D1;
  float* y1b  = ws + off; off += (size_t)Ny * D1;
  float* Px2a = ws + off; off += (size_t)Nx * C;
  float* Px2b = ws + off; off += (size_t)Ny * C;
  float* Py2a = ws + off; off += (size_t)Ny * C;
  float* Py2b = ws + off; off += (size_t)Nx * C;
  float* x2b  = ws + off; off += (size_t)Nx * C;
  float* y2b  = ws + off; off += (size_t)Ny * C;

  float* hhat = (float*)d_out;                       // 8192x4096
  float* xout = (float*)d_out + (size_t)Nx * Ny;     // 8192x64

  dim3 blk(NTHREADS);
  const float* NZ = nullptr;

  // ---- stage 1: feature projections  P = inp @ W ----
  hcon_dual_gemm<<<dim3(D1/TILE_N, Nx/TILE_M), blk, 0, stream>>>(
      x0, Fx, Wx1a, D1, Fx,  NZ,0,NZ,0,0,  NZ,NZ, NZ,1.f,0.f, Px1a, D1, 0, 0);
  hcon_dual_gemm<<<dim3(D1/TILE_N, Ny/TILE_M), blk, 0, stream>>>(
      y0, Fy, Wx1b, D1, Fy,  NZ,0,NZ,0,0,  NZ,NZ, NZ,1.f,0.f, Px1b, D1, 0, 0);
  hcon_dual_gemm<<<dim3(D1/TILE_N, Ny/TILE_M), blk, 0, stream>>>(
      y0, Fy, Wy1a, D1, Fy,  NZ,0,NZ,0,0,  NZ,NZ, NZ,1.f,0.f, Py1a, D1, 0, 0);
  hcon_dual_gemm<<<dim3(D1/TILE_N, Nx/TILE_M), blk, 0, stream>>>(
      x0, Fx, Wy1b, D1, Fx,  NZ,0,NZ,0,0,  NZ,NZ, NZ,1.f,0.f, Py1b, D1, 0, 0);

  // ---- layer 1 combine ----
  hcon_dual_gemm<<<dim3(D1/TILE_N, Nx/TILE_M), blk, 0, stream>>>(
      hx1, Nx, Px1a, D1, Nx,  hx2, Ny, Px1b, D1, Ny,
      bx1a, bx1b, alpha, 0.f, 0.f, x1b, D1, 0, 1);
  hcon_dual_gemm<<<dim3(D1/TILE_N, Ny/TILE_M), blk, 0, stream>>>(
      hy1, Ny, Py1a, D1, Ny,  hy2, Nx, Py1b, D1, Nx,
      by1a, by1b, beta, 0.f, 0.f, y1b, D1, 0, 1);

  // ---- stage 2 projections ----
  hcon_dual_gemm<<<dim3(C/TILE_N, Nx/TILE_M), blk, 0, stream>>>(
      x1b, D1, Wx2a, C, D1,  NZ,0,NZ,0,0,  NZ,NZ, NZ,1.f,0.f, Px2a, C, 0, 0);
  hcon_dual_gemm<<<dim3(C/TILE_N, Ny/TILE_M), blk, 0, stream>>>(
      y1b, D1, Wx2b, C, D1,  NZ,0,NZ,0,0,  NZ,NZ, NZ,1.f,0.f, Px2b, C, 0, 0);
  hcon_dual_gemm<<<dim3(C/TILE_N, Ny/TILE_M), blk, 0, stream>>>(
      y1b, D1, Wy2a, C, D1,  NZ,0,NZ,0,0,  NZ,NZ, NZ,1.f,0.f, Py2a, C, 0, 0);
  hcon_dual_gemm<<<dim3(C/TILE_N, Nx/TILE_M), blk, 0, stream>>>(
      x1b, D1, Wy2b, C, D1,  NZ,0,NZ,0,0,  NZ,NZ, NZ,1.f,0.f, Py2b, C, 0, 0);

  // ---- layer 2 combine ----
  hcon_dual_gemm<<<dim3(C/TILE_N, Nx/TILE_M), blk, 0, stream>>>(
      hx1, Nx, Px2a, C, Nx,  hx2, Ny, Px2b, C, Ny,
      bx2a, bx2b, alpha, 0.f, 0.f, x2b, C, 0, 1);
  hcon_dual_gemm<<<dim3(C/TILE_N, Ny/TILE_M), blk, 0, stream>>>(
      hy1, Ny, Py2a, C, Ny,  hy2, Nx, Py2b, C, Nx,
      by2a, by2b, beta, 0.f, 0.f, y2b, C, 0, 1);

  // ---- h_hat = sigmoid(x2 @ y2^T) ----
  hcon_dual_gemm<<<dim3(Ny/TILE_N, Nx/TILE_M), blk, 0, stream>>>(
      x2b, C, y2b, C, C,  NZ,0,NZ,0,0,  NZ,NZ, NZ,1.f,0.f, hhat, Ny, 1, 2);

  // ---- x_output = log_softmax(x2) ----
  hcon_log_softmax64<<<dim3(Nx/8), dim3(256), 0, stream>>>(x2b, xout);
}
